// OAPaiNN_63934883168409
// MI455X (gfx1250) — matrix operand order repaired and checked by reading, not compiled, not run on hardware
//
#include <hip/hip_runtime.h>
#include <math.h>

typedef __bf16 bf16_t;
typedef __attribute__((ext_vector_type(16))) __bf16 v16bf;
typedef __attribute__((ext_vector_type(8)))  float   v8f;

#define INV3f 0.57735026918962576f
#define INV2f 0.70710678118654752f

__device__ __forceinline__ float ssilu_f(float x) {
  return x * (1.0f / 0.6f) / (1.0f + __expf(-x));
}

// CDNA5 async global->LDS copy (ASYNCcnt tracked), 16 bytes per lane.
__device__ __forceinline__ void async_load_b128(unsigned lds_off, const void* gptr) {
  asm volatile("global_load_async_to_lds_b128 %0, %1, off"
               :: "v"(lds_off), "v"(gptr) : "memory");
}
__device__ __forceinline__ void wait_async_all() {
  asm volatile("s_wait_asynccnt 0x0" ::: "memory");
}

// ---------------------------------------------------------------------------
// Generic bf16 WMMA GEMM: out = scale * (act(A[M,K] * W[N,K]^T + bias) [+ res])
//   block tile 128(M) x 128(N), K step 32, 8 waves (4x2), wave tile 32x64.
//   Double-buffered LDS fed by global_load_async_to_lds_b128; one barrier/step.
// ---------------------------------------------------------------------------
__global__ __launch_bounds__(256)
void wmma_gemm_k(const bf16_t* __restrict__ A, int lda,
                 const bf16_t* __restrict__ W,              // [Nn][K] row-major
                 const float*  __restrict__ bias,           // [Nn] or null
                 const bf16_t* __restrict__ resB, int ldresB,
                 const float*  __restrict__ resF, int ldresF,
                 float scale, int act,
                 float* __restrict__ outF, bf16_t* __restrict__ outB, int ldo,
                 int M, int Nn, int K)
{
  __shared__ __align__(16) bf16_t As[2][128 * 40];   // 32-col K tile, stride 40 (pad)
  __shared__ __align__(16) bf16_t Bs[2][128 * 40];

  const int t    = threadIdx.x;
  const int lane = t & 31;
  const int wave = t >> 5;
  const int wm   = (wave >> 1) * 32;   // wave M offset in block tile (4 waves)
  const int wn   = (wave & 1) * 64;    // wave N offset in block tile (2 waves)
  const int blockM = blockIdx.x * 128;
  const int blockN = blockIdx.y * 128;

  v8f acc[2][4];
  #pragma unroll
  for (int i = 0; i < 2; ++i)
    #pragma unroll
    for (int j = 0; j < 4; ++j)
      #pragma unroll
      for (int q = 0; q < 8; ++q) acc[i][j][q] = 0.0f;

  const int nk   = K >> 5;
  const int arow = t >> 1, aseg = (t & 1) * 16;   // 128 rows x 32 cols, 32B/thread
  const int brow = t >> 1, bseg = (t & 1) * 16;
  const int gm   = blockM + arow;
  const int gn   = blockN + brow;
  const int lrow = lane & 15;
  const int khalf = lane >> 4;

  // LDS byte addresses (low 32 bits of flat shared pointer = LDS offset)
  const unsigned ldsA[2] = {
    (unsigned)(unsigned long long)&As[0][arow * 40 + aseg],
    (unsigned)(unsigned long long)&As[1][arow * 40 + aseg] };
  const unsigned ldsB[2] = {
    (unsigned)(unsigned long long)&Bs[0][brow * 40 + bseg],
    (unsigned)(unsigned long long)&Bs[1][brow * 40 + bseg] };

  // Issue async loads of K-tile kt into LDS buffer buf. OOB lanes skip their
  // copy; the stale LDS rows only feed outputs masked off in the epilogue.
  auto issue_tile = [&](int kt, int buf) {
    const int k0 = kt << 5;
    if (gm < M) {
      const char* g = (const char*)(A + (long long)gm * lda + k0 + aseg);
      async_load_b128(ldsA[buf],      g);
      async_load_b128(ldsA[buf] + 16, g + 16);
    }
    if (gn < Nn) {
      const char* g = (const char*)(W + (long long)gn * K + k0 + bseg);
      async_load_b128(ldsB[buf],      g);
      async_load_b128(ldsB[buf] + 16, g + 16);
    }
  };

  issue_tile(0, 0);
  wait_async_all();
  __syncthreads();

  for (int kt = 0; kt < nk; ++kt) {
    const int cur = kt & 1;
    if (kt + 1 < nk) issue_tile(kt + 1, cur ^ 1);

    // A fragments: lane L -> row (L&15); K halves 0-7/16-23 (lanes 0-15)
    // or 8-15/24-31 (lanes 16-31), packed 2 per VGPR.
    v16bf af[2], bfrag[4];
    #pragma unroll
    for (int i = 0; i < 2; ++i) {
      const bf16_t* p = &As[cur][(wm + i * 16 + lrow) * 40 + khalf * 8];
      v16bf f;
      #pragma unroll
      for (int q = 0; q < 8; ++q) { f[q] = p[q]; f[q + 8] = p[16 + q]; }
      af[i] = f;
    }
    // B fragments: lane L -> column (L&15); K 0-7/16-23 (lanes 0-15) / 8-15/24-31.
    const int kb = khalf * 8;
    #pragma unroll
    for (int j = 0; j < 4; ++j) {
      const bf16_t* p = &Bs[cur][(wn + j * 16 + lrow) * 40 + kb];
      v16bf f;
      #pragma unroll
      for (int q = 0; q < 8; ++q) { f[q] = p[q]; f[q + 8] = p[16 + q]; }
      bfrag[j] = f;
    }

    #pragma unroll
    for (int i = 0; i < 2; ++i)
      #pragma unroll
      for (int j = 0; j < 4; ++j)
        acc[i][j] = __builtin_amdgcn_wmma_f32_16x16x32_bf16(
            false, af[i], false, bfrag[j], (short)0, acc[i][j], false, false);

    wait_async_all();   // next tile landed in LDS
    __syncthreads();    // all waves done reading cur, next buf valid for all
  }

  // Epilogue: C/D layout -> element r of lane L is row (L>>4)*8+r, col L&15.
  const int rbase = (lane >> 4) * 8;
  const int col   = lane & 15;
  #pragma unroll
  for (int i = 0; i < 2; ++i) {
    #pragma unroll
    for (int j = 0; j < 4; ++j) {
      const int mBase = blockM + wm + i * 16;
      const int nBase = blockN + wn + j * 16;
      #pragma unroll
      for (int r = 0; r < 8; ++r) {
        const int m = mBase + rbase + r;
        const int n = nBase + col;
        if (m < M && n < Nn) {
          float v = acc[i][j][r];
          if (bias) v += bias[n];
          if (act == 1) v = ssilu_f(v);
          if (resB) v += (float)resB[(long long)m * ldresB + n];
          if (resF) v += resF[(long long)m * ldresF + n];
          v *= scale;
          if (outB) outB[(long long)m * ldo + n] = (bf16_t)v;
          else      outF[(long long)m * ldo + n] = v;
        }
      }
    }
  }
}

// ---------------------------------------------------------------------------
// Support kernels
// ---------------------------------------------------------------------------
__global__ void f32_to_bf16_k(const float* __restrict__ in, bf16_t* __restrict__ out, long long n) {
  long long id = (long long)blockIdx.x * 256 + threadIdx.x;
  if (id < n) out[id] = (bf16_t)in[id];
}

__global__ void zero_f32_k(float* __restrict__ p, long long n) {
  long long id = (long long)blockIdx.x * 256 + threadIdx.x;
  if (id < n) p[id] = 0.0f;
}

__global__ void edge_geom_k(const float* __restrict__ pos, const int* __restrict__ src,
                            const int* __restrict__ dst, float* __restrict__ dist,
                            float* __restrict__ unit, int E) {
  int e = blockIdx.x * 256 + threadIdx.x;
  if (e >= E) return;
  int s = src[e], d = dst[e];
  float dx = pos[s * 3 + 0] - pos[d * 3 + 0];
  float dy = pos[s * 3 + 1] - pos[d * 3 + 1];
  float dz = pos[s * 3 + 2] - pos[d * 3 + 2];
  float r = sqrtf(dx * dx + dy * dy + dz * dz);
  dist[e] = r;
  float inv = 1.0f / fmaxf(r, 1e-9f);
  unit[e * 3 + 0] = dx * inv;
  unit[e * 3 + 1] = dy * inv;
  unit[e * 3 + 2] = dz * inv;
}

__global__ void rbf_k(const float* __restrict__ dist, bf16_t* __restrict__ rbf, int E) {
  int e = blockIdx.x;            // grid = E blocks of 128 threads
  int r = threadIdx.x;
  float d = dist[e] * 0.1f;      // /CUTOFF
  float env = 0.0f;
  if (d < 1.0f) {
    float d5 = d * d * d * d * d;
    env = 1.0f - 21.0f * d5 + 35.0f * d5 * d - 15.0f * d5 * d * d;  // p=5 envelope
  }
  float off = (float)r * (1.0f / 127.0f);
  float tt = d - off;
  float coeff = -0.5f * 127.0f * 127.0f;
  rbf[(long long)e * 128 + r] = (bf16_t)(env * __expf(coeff * tt * tt));
}

__global__ void count_edges_k(const int* __restrict__ dst, float* __restrict__ cnt, int E) {
  int e = blockIdx.x * 256 + threadIdx.x;
  if (e < E) atomicAdd(&cnt[dst[e]], 1.0f);
}

__global__ __launch_bounds__(256)
void layernorm_k(const float* __restrict__ x, const float* __restrict__ g,
                 const float* __restrict__ b, bf16_t* __restrict__ out, int N, int C) {
  int wave = threadIdx.x >> 5, lane = threadIdx.x & 31;
  int row = blockIdx.x * 8 + wave;
  if (row >= N) return;
  const float* xr = x + (long long)row * C;
  float s = 0.0f;
  for (int c = lane; c < C; c += 32) s += xr[c];
  #pragma unroll
  for (int off = 16; off > 0; off >>= 1) s += __shfl_xor(s, off, 32);
  float mu = s / (float)C;
  float v = 0.0f;
  for (int c = lane; c < C; c += 32) { float dd = xr[c] - mu; v += dd * dd; }
  #pragma unroll
  for (int off = 16; off > 0; off >>= 1) v += __shfl_xor(v, off, 32);
  float rstd = rsqrtf(v / (float)C + 1e-5f);
  for (int c = lane; c < C; c += 32)
    out[(long long)row * C + c] = (bf16_t)((xr[c] - mu) * rstd * g[c] + b[c]);
}

__global__ void gather_mij_k(const bf16_t* __restrict__ xn, const int* __restrict__ src,
                             const int* __restrict__ dst, bf16_t* __restrict__ mij, int E) {
  int e = blockIdx.x, c = threadIdx.x;            // grid E x 256
  mij[(long long)e * 768 + c]       = xn[(long long)dst[e] * 256 + c];
  mij[(long long)e * 768 + 256 + c] = xn[(long long)src[e] * 256 + c];
}

__global__ void scatter_mean_k(const bf16_t* __restrict__ mij, const int* __restrict__ dst,
                               float* __restrict__ acc, int E) {
  long long id = (long long)blockIdx.x * 256 + threadIdx.x;
  if (id >= (long long)E * 768) return;
  int e = (int)(id / 768), c = (int)(id % 768);
  atomicAdd(&acc[(long long)dst[e] * 768 + c], (float)mij[id]);
}

__global__ void mean_div_k(const float* __restrict__ acc, const float* __restrict__ cnt,
                           bf16_t* __restrict__ out, int N) {
  long long id = (long long)blockIdx.x * 256 + threadIdx.x;
  if (id >= (long long)N * 768) return;
  int n = (int)(id / 768);
  out[id] = (bf16_t)(acc[id] / fmaxf(cnt[n], 1.0f));
}

__global__ void pm_scatter_k(const bf16_t* __restrict__ xp, const bf16_t* __restrict__ rbfh,
                             const float* __restrict__ vec, const float* __restrict__ unit,
                             const int* __restrict__ src, const int* __restrict__ dst,
                             const int* __restrict__ mask, float* __restrict__ dx,
                             float* __restrict__ dvec, int E) {
  int e = blockIdx.x, c = threadIdx.x;            // grid E x 256
  if (mask[e] == 0) return;
  int s = src[e], d = dst[e];
  float x1 = (float)xp[(long long)s * 768 + c]       * (float)rbfh[(long long)e * 768 + c];
  float x2 = (float)xp[(long long)s * 768 + 256 + c] * (float)rbfh[(long long)e * 768 + 256 + c] * INV3f;
  float x3 = (float)xp[(long long)s * 768 + 512 + c] * (float)rbfh[(long long)e * 768 + 512 + c];
  atomicAdd(&dx[(long long)d * 256 + c], x1);
  #pragma unroll
  for (int dd = 0; dd < 3; ++dd) {
    float vm = (vec[(long long)s * 768 + dd * 256 + c] * x2 + x3 * unit[e * 3 + dd]) * 0.0625f; // invH
    atomicAdd(&dvec[(long long)d * 768 + dd * 256 + c], vm);
  }
}

__global__ void pm_update_x_k(float* __restrict__ x, const float* __restrict__ dx, long long n) {
  long long id = (long long)blockIdx.x * 256 + threadIdx.x;
  if (id < n) x[id] = (x[id] + dx[id]) * INV2f;
}

__global__ void add_inplace_k(float* __restrict__ a, const float* __restrict__ b, long long n) {
  long long id = (long long)blockIdx.x * 256 + threadIdx.x;
  if (id < n) a[id] += b[id];
}

__global__ void vdot_vn_k(const float* __restrict__ vp, float* __restrict__ vdot,
                          float* __restrict__ vn, int N) {
  long long id = (long long)blockIdx.x * 256 + threadIdx.x;
  if (id >= (long long)N * 256) return;
  int n = (int)(id >> 8), c = (int)(id & 255);
  float s1 = 0.0f, s2 = 0.0f;
  #pragma unroll
  for (int d = 0; d < 3; ++d) {
    float a = vp[((long long)n * 3 + d) * 512 + c];
    float b = vp[((long long)n * 3 + d) * 512 + 256 + c];
    s1 += a * b; s2 += b * b;
  }
  vdot[id] = s1 * 0.0625f;            // invH = 1/sqrt(256)
  vn[id]   = sqrtf(s2 + 1e-8f);
}

__global__ void concat2_bf16_k(const float* __restrict__ a, int Ca,
                               const float* __restrict__ b, int Cb,
                               bf16_t* __restrict__ out, int N) {
  int C = Ca + Cb;
  long long id = (long long)blockIdx.x * 256 + threadIdx.x;
  if (id >= (long long)N * C) return;
  int n = (int)(id / C), c = (int)(id % C);
  float v = (c < Ca) ? a[(long long)n * Ca + c] : b[(long long)n * Cb + (c - Ca)];
  out[id] = (bf16_t)v;
}

__global__ void up_update_k(float* __restrict__ x, float* __restrict__ vec,
                            const float* __restrict__ u, const float* __restrict__ vdot,
                            const float* __restrict__ vp, int N) {
  long long id = (long long)blockIdx.x * 256 + threadIdx.x;
  if (id >= (long long)N * 256) return;
  int n = (int)(id >> 8), c = (int)(id & 255);
  float a  = u[(long long)n * 768 + c];
  float bg = u[(long long)n * 768 + 256 + c];
  float cc = u[(long long)n * 768 + 512 + c];
  x[id] = (x[id] + (a + bg * vdot[id]) * INV2f) * INV2f;
  #pragma unroll
  for (int d = 0; d < 3; ++d)
    vec[(long long)n * 768 + d * 256 + c] += cc * vp[((long long)n * 3 + d) * 512 + c]; // v1
}

__global__ void norm3_k(const float* __restrict__ vp, float* __restrict__ out, int N, int C) {
  long long id = (long long)blockIdx.x * 256 + threadIdx.x;
  if (id >= (long long)N * C) return;
  int n = (int)(id / C), c = (int)(id % C);
  float s = 0.0f;
  #pragma unroll
  for (int d = 0; d < 3; ++d) {
    float v = vp[((long long)n * 3 + d) * C + c];
    s += v * v;
  }
  out[id] = sqrtf(s);
}

__global__ void geb1_combine_k(const float* __restrict__ u, const float* __restrict__ vpre,
                               float* __restrict__ xo, bf16_t* __restrict__ vob, int N) {
  long long id = (long long)blockIdx.x * 256 + threadIdx.x;
  if (id >= (long long)N * 128) return;
  int n = (int)(id >> 7), c = (int)(id & 127);
  xo[id] = ssilu_f(u[(long long)n * 256 + c]);
  float g = u[(long long)n * 256 + 128 + c];
  #pragma unroll
  for (int d = 0; d < 3; ++d)
    vob[((long long)n * 3 + d) * 128 + c] = (bf16_t)(g * vpre[((long long)n * 3 + d) * 128 + c]);
}

__global__ void final_vo_k(const float* __restrict__ u2, const float* __restrict__ vec2b,
                           float* __restrict__ out, int N) {
  int i = blockIdx.x * 256 + threadIdx.x;
  if (i >= N * 3) return;
  int n = i / 3;
  out[i] = u2[n * 2 + 1] * vec2b[i];
}

__global__ void naive_gemm_f32A_k(const float* __restrict__ A, int lda,
                                  const float* __restrict__ W, const float* __restrict__ bias,
                                  float* __restrict__ out, int ldo,
                                  int M, int Nn, int K, int act) {
  long long id = (long long)blockIdx.x * 256 + threadIdx.x;
  if (id >= (long long)M * Nn) return;
  int m = (int)(id / Nn), n = (int)(id % Nn);
  float s = bias ? bias[n] : 0.0f;
  const float* ar = A + (long long)m * lda;
  const float* wr = W + (long long)n * K;
  for (int k = 0; k < K; ++k) s += ar[k] * wr[k];
  if (act) s = ssilu_f(s);
  out[(long long)m * ldo + n] = s;
}

__global__ void naive_gemm_bf16A_k(const bf16_t* __restrict__ A, int lda,
                                   const float* __restrict__ W, const float* __restrict__ bias,
                                   float* __restrict__ out, int ldo,
                                   int M, int Nn, int K, int act) {
  long long id = (long long)blockIdx.x * 256 + threadIdx.x;
  if (id >= (long long)M * Nn) return;
  int m = (int)(id / Nn), n = (int)(id % Nn);
  float s = bias ? bias[n] : 0.0f;
  const bf16_t* ar = A + (long long)m * lda;
  const float* wr = W + (long long)n * K;
  for (int k = 0; k < K; ++k) s += (float)ar[k] * wr[k];
  if (act) s = ssilu_f(s);
  out[(long long)m * ldo + n] = s;
}

// ---------------------------------------------------------------------------
// Host orchestration
// ---------------------------------------------------------------------------
static void launch_gemm(hipStream_t s, const bf16_t* A, int lda, const bf16_t* W,
                        const float* bias, const bf16_t* resB, int ldresB,
                        const float* resF, int ldresF, float scale, int act,
                        float* outF, bf16_t* outB, int ldo, int M, int Nn, int K) {
  dim3 g((unsigned)((M + 127) / 128), (unsigned)((Nn + 127) / 128));
  wmma_gemm_k<<<g, 256, 0, s>>>(A, lda, W, bias, resB, ldresB, resF, ldresF,
                                scale, act, outF, outB, ldo, M, Nn, K);
}

extern "C" void kernel_launch(void* const* d_in, const int* in_sizes, int n_in,
                              void* d_out, int out_size, void* d_ws, size_t ws_size,
                              hipStream_t stream) {
  const float* xh       = (const float*)d_in[0];
  const float* pos      = (const float*)d_in[1];
  const int*   ei       = (const int*)d_in[2];
  const int*   mask     = (const int*)d_in[3];
  const float* emb_W    = (const float*)d_in[4];
  const float* emb_b    = (const float*)d_in[5];
  const float* gm_ln_g  = (const float*)d_in[6];
  const float* gm_ln_b  = (const float*)d_in[7];
  const float* gm_rbf_W = (const float*)d_in[8];
  const float* gm_rbf_b = (const float*)d_in[9];
  const float* gm_eW1   = (const float*)d_in[10];
  const float* gm_eb1   = (const float*)d_in[11];
  const float* gm_eW2   = (const float*)d_in[12];
  const float* gm_eb2   = (const float*)d_in[13];
  const float* gm_nW1   = (const float*)d_in[14];
  const float* gm_nb1   = (const float*)d_in[15];
  const float* gm_nW2   = (const float*)d_in[16];
  const float* gm_nb2   = (const float*)d_in[17];
  const float* pm_ln_g  = (const float*)d_in[18];
  const float* pm_ln_b  = (const float*)d_in[19];
  const float* pm_xW1   = (const float*)d_in[20];
  const float* pm_xb1   = (const float*)d_in[21];
  const float* pm_xW2   = (const float*)d_in[22];
  const float* pm_xb2   = (const float*)d_in[23];
  const float* pm_rbf_W = (const float*)d_in[24];
  const float* pm_rbf_b = (const float*)d_in[25];
  const float* up_vW    = (const float*)d_in[26];
  const float* up_xW1   = (const float*)d_in[27];
  const float* up_xb1   = (const float*)d_in[28];
  const float* up_xW2   = (const float*)d_in[29];
  const float* up_xb2   = (const float*)d_in[30];
  const float* ox_W1    = (const float*)d_in[31];
  const float* ox_b1    = (const float*)d_in[32];
  const float* ox_W2    = (const float*)d_in[33];
  const float* ox_b2    = (const float*)d_in[34];
  const float* o1_v1W   = (const float*)d_in[35];
  const float* o1_v2W   = (const float*)d_in[36];
  const float* o1_uW1   = (const float*)d_in[37];
  const float* o1_ub1   = (const float*)d_in[38];
  const float* o1_uW2   = (const float*)d_in[39];
  const float* o1_ub2   = (const float*)d_in[40];
  const float* o2_v1W   = (const float*)d_in[41];
  const float* o2_v2W   = (const float*)d_in[42];
  const float* o2_uW1   = (const float*)d_in[43];
  const float* o2_ub1   = (const float*)d_in[44];
  const float* o2_uW2   = (const float*)d_in[45];
  const float* o2_ub2   = (const float*)d_in[46];

  const int N = in_sizes[1] / 3;
  const int E = in_sizes[3];
  const int L = in_sizes[6] / 256;
  const int H = 256, R = 128;
  const int* src = ei;
  const int* dst = ei + E;
  float* outp = (float*)d_out;

  // bump allocator over workspace
  char* wp = (char*)d_ws;
  auto alloc = [&](size_t bytes) -> void* {
    void* p = (void*)wp;
    wp += (bytes + 255) & ~(size_t)255;
    return p;
  };
  auto g1 = [](long long n) { return dim3((unsigned)((n + 255) / 256)); };

  float* dist    = (float*)alloc(sizeof(float) * E);
  float* unit    = (float*)alloc(sizeof(float) * E * 3);
  float* cnt     = (float*)alloc(sizeof(float) * N);
  float* x       = (float*)alloc(sizeof(float) * (size_t)N * H);
  float* vec     = (float*)alloc(sizeof(float) * (size_t)N * 3 * H);
  float* dx      = (float*)alloc(sizeof(float) * (size_t)N * H);
  float* dvec    = (float*)alloc(sizeof(float) * (size_t)N * 3 * H);
  float* nodeacc = (float*)alloc(sizeof(float) * (size_t)N * 3 * H);
  float* vp      = (float*)alloc(sizeof(float) * (size_t)N * 3 * 512);
  float* vdot    = (float*)alloc(sizeof(float) * (size_t)N * H);
  float* vnb     = (float*)alloc(sizeof(float) * (size_t)N * H);
  float* ubuf    = (float*)alloc(sizeof(float) * (size_t)N * 3 * H);
  float* xo      = (float*)alloc(sizeof(float) * (size_t)N * 128);
  float* vpre    = (float*)alloc(sizeof(float) * (size_t)N * 3 * 128);
  float* vec2b   = (float*)alloc(sizeof(float) * (size_t)N * 3);
  bf16_t* rbf    = (bf16_t*)alloc(sizeof(bf16_t) * (size_t)E * R);
  bf16_t* mij    = (bf16_t*)alloc(sizeof(bf16_t) * (size_t)E * 768);
  bf16_t* hid    = (bf16_t*)alloc(sizeof(bf16_t) * (size_t)E * 768);
  bf16_t* xn     = (bf16_t*)alloc(sizeof(bf16_t) * (size_t)N * H);
  bf16_t* meanb  = (bf16_t*)alloc(sizeof(bf16_t) * (size_t)N * 768);
  bf16_t* xp     = (bf16_t*)alloc(sizeof(bf16_t) * (size_t)N * 768);
  bf16_t* vecb   = (bf16_t*)alloc(sizeof(bf16_t) * (size_t)N * 768);
  bf16_t* xcat   = (bf16_t*)alloc(sizeof(bf16_t) * (size_t)N * 512);
  bf16_t* vob    = (bf16_t*)alloc(sizeof(bf16_t) * (size_t)N * 3 * 128);

  auto convW = [&](const float* w, size_t n) -> bf16_t* {
    bf16_t* p = (bf16_t*)alloc(sizeof(bf16_t) * n);
    f32_to_bf16_k<<<dim3((unsigned)((n + 255) / 256)), 256, 0, stream>>>(w, p, (long long)n);
    return p;
  };

  bf16_t* b_gm_rbf = convW(gm_rbf_W, (size_t)L * H * R);
  bf16_t* b_eW1  = convW(gm_eW1, (size_t)L * 768 * 768);
  bf16_t* b_eW2  = convW(gm_eW2, (size_t)L * 768 * 768);
  bf16_t* b_nW1  = convW(gm_nW1, (size_t)L * 512 * 768);
  bf16_t* b_nW2  = convW(gm_nW2, (size_t)L * 256 * 512);
  bf16_t* b_pxW1 = convW(pm_xW1, (size_t)L * 256 * 256);
  bf16_t* b_pxW2 = convW(pm_xW2, (size_t)L * 768 * 256);
  bf16_t* b_prbf = convW(pm_rbf_W, (size_t)L * 768 * 128);
  bf16_t* b_uvW  = convW(up_vW,  (size_t)L * 512 * 256);
  bf16_t* b_uxW1 = convW(up_xW1, (size_t)L * 256 * 512);
  bf16_t* b_uxW2 = convW(up_xW2, (size_t)L * 768 * 256);
  bf16_t* b_oxW1 = convW(ox_W1, (size_t)128 * 256);
  bf16_t* b_o1v1 = convW(o1_v1W, (size_t)256 * 256);
  bf16_t* b_o1v2 = convW(o1_v2W, (size_t)128 * 256);
  bf16_t* b_o1u1 = convW(o1_uW1, (size_t)256 * 512);
  bf16_t* b_o1u2 = convW(o1_uW2, (size_t)256 * 256);
  bf16_t* b_o2v1 = convW(o2_v1W, (size_t)128 * 128);
  bf16_t* b_o2u1 = convW(o2_uW1, (size_t)128 * 256);

  // --- setup ---
  edge_geom_k<<<g1(E), 256, 0, stream>>>(pos, src, dst, dist, unit, E);
  rbf_k<<<E, 128, 0, stream>>>(dist, rbf, E);
  zero_f32_k<<<g1(N), 256, 0, stream>>>(cnt, N);
  count_edges_k<<<g1(E), 256, 0, stream>>>(dst, cnt, E);
  naive_gemm_f32A_k<<<g1((long long)N * H), 256, 0, stream>>>(xh, 8, emb_W, emb_b, x, H, N, H, 8, 0);
  zero_f32_k<<<g1((long long)N * 768), 256, 0, stream>>>(vec, (long long)N * 768);

  // --- layers ---
  for (int l = 0; l < L; ++l) {
    // graph-message block
    layernorm_k<<<dim3((unsigned)((N + 7) / 8)), 256, 0, stream>>>(x, gm_ln_g + l * H, gm_ln_b + l * H, xn, N, H);
    launch_gemm(stream, rbf, R, b_gm_rbf + (size_t)l * H * R, gm_rbf_b + l * H,
                nullptr, 0, nullptr, 0, 1.0f, 0, nullptr, mij + 512, 768, E, H, R);
    gather_mij_k<<<E, 256, 0, stream>>>(xn, src, dst, mij, E);
    launch_gemm(stream, mij, 768, b_eW1 + (size_t)l * 768 * 768, gm_eb1 + l * 768,
                nullptr, 0, nullptr, 0, 1.0f, 1, nullptr, hid, 768, E, 768, 768);
    launch_gemm(stream, hid, 768, b_eW2 + (size_t)l * 768 * 768, gm_eb2 + l * 768,
                mij, 768, nullptr, 0, INV3f, 0, nullptr, mij, 768, E, 768, 768);
    zero_f32_k<<<g1((long long)N * 768), 256, 0, stream>>>(nodeacc, (long long)N * 768);
    scatter_mean_k<<<g1((long long)E * 768), 256, 0, stream>>>(mij, dst, nodeacc, E);
    mean_div_k<<<g1((long long)N * 768), 256, 0, stream>>>(nodeacc, cnt, meanb, N);
    launch_gemm(stream, meanb, 768, b_nW1 + (size_t)l * 512 * 768, gm_nb1 + l * 512,
                nullptr, 0, nullptr, 0, 1.0f, 1, nullptr, hid, 512, N, 512, 768);
    launch_gemm(stream, hid, 512, b_nW2 + (size_t)l * 256 * 512, gm_nb2 + l * 256,
                nullptr, 0, x, 256, 1.0f, 0, x, nullptr, 256, N, 256, 512);

    // position-message block
    layernorm_k<<<dim3((unsigned)((N + 7) / 8)), 256, 0, stream>>>(x, pm_ln_g + l * H, pm_ln_b + l * H, xn, N, H);
    launch_gemm(stream, xn, 256, b_pxW1 + (size_t)l * 256 * 256, pm_xb1 + l * 256,
                nullptr, 0, nullptr, 0, 1.0f, 1, nullptr, hid, 256, N, 256, 256);
    launch_gemm(stream, hid, 256, b_pxW2 + (size_t)l * 768 * 256, pm_xb2 + l * 768,
                nullptr, 0, nullptr, 0, 1.0f, 0, nullptr, xp, 768, N, 768, 256);
    launch_gemm(stream, rbf, R, b_prbf + (size_t)l * 768 * 128, pm_rbf_b + l * 768,
                nullptr, 0, nullptr, 0, 1.0f, 0, nullptr, hid, 768, E, 768, 128);
    zero_f32_k<<<g1((long long)N * 256), 256, 0, stream>>>(dx, (long long)N * 256);
    zero_f32_k<<<g1((long long)N * 768), 256, 0, stream>>>(dvec, (long long)N * 768);
    pm_scatter_k<<<E, 256, 0, stream>>>(xp, hid, vec, unit, src, dst, mask, dx, dvec, E);
    pm_update_x_k<<<g1((long long)N * 256), 256, 0, stream>>>(x, dx, (long long)N * 256);
    add_inplace_k<<<g1((long long)N * 768), 256, 0, stream>>>(vec, dvec, (long long)N * 768);

    // update block
    f32_to_bf16_k<<<g1((long long)N * 768), 256, 0, stream>>>(vec, vecb, (long long)N * 768);
    launch_gemm(stream, vecb, 256, b_uvW + (size_t)l * 512 * 256, nullptr,
                nullptr, 0, nullptr, 0, 1.0f, 0, vp, nullptr, 512, N * 3, 512, 256);
    vdot_vn_k<<<g1((long long)N * 256), 256, 0, stream>>>(vp, vdot, vnb, N);
    concat2_bf16_k<<<g1((long long)N * 512), 256, 0, stream>>>(x, 256, vnb, 256, xcat, N);
    launch_gemm(stream, xcat, 512, b_uxW1 + (size_t)l * 256 * 512, up_xb1 + l * 256,
                nullptr, 0, nullptr, 0, 1.0f, 1, nullptr, hid, 256, N, 256, 512);
    launch_gemm(stream, hid, 256, b_uxW2 + (size_t)l * 768 * 256, up_xb2 + l * 768,
                nullptr, 0, nullptr, 0, 1.0f, 0, ubuf, nullptr, 768, N, 768, 256);
    up_update_k<<<g1((long long)N * 256), 256, 0, stream>>>(x, vec, ubuf, vdot, vp, N);
  }

  // --- output heads ---
  f32_to_bf16_k<<<g1((long long)N * 256), 256, 0, stream>>>(x, xn, (long long)N * 256);
  launch_gemm(stream, xn, 256, b_oxW1, ox_b1, nullptr, 0, nullptr, 0, 1.0f, 1,
              nullptr, hid, 128, N, 128, 256);
  naive_gemm_bf16A_k<<<g1((long long)N * 8), 256, 0, stream>>>(hid, 128, ox_W2, ox_b2, outp, 8, N, 8, 128, 0);

  // geb 1
  f32_to_bf16_k<<<g1((long long)N * 768), 256, 0, stream>>>(vec, vecb, (long long)N * 768);
  launch_gemm(stream, vecb, 256, b_o1v1, nullptr, nullptr, 0, nullptr, 0, 1.0f, 0,
              vp, nullptr, 256, N * 3, 256, 256);
  launch_gemm(stream, vecb, 256, b_o1v2, nullptr, nullptr, 0, nullptr, 0, 1.0f, 0,
              vpre, nullptr, 128, N * 3, 128, 256);
  norm3_k<<<g1((long long)N * 256), 256, 0, stream>>>(vp, vnb, N, 256);
  concat2_bf16_k<<<g1((long long)N * 512), 256, 0, stream>>>(x, 256, vnb, 256, xcat, N);
  launch_gemm(stream, xcat, 512, b_o1u1, o1_ub1, nullptr, 0, nullptr, 0, 1.0f, 1,
              nullptr, hid, 256, N, 256, 512);
  launch_gemm(stream, hid, 256, b_o1u2, o1_ub2, nullptr, 0, nullptr, 0, 1.0f, 0,
              ubuf, nullptr, 256, N, 256, 256);
  geb1_combine_k<<<g1((long long)N * 128), 256, 0, stream>>>(ubuf, vpre, xo, vob, N);

  // geb 2
  launch_gemm(stream, vob, 128, b_o2v1, nullptr, nullptr, 0, nullptr, 0, 1.0f, 0,
              vp, nullptr, 128, N * 3, 128, 128);
  naive_gemm_bf16A_k<<<g1((long long)N * 3), 256, 0, stream>>>(vob, 128, o2_v2W, nullptr, vec2b, 1, N * 3, 1, 128, 0);
  norm3_k<<<g1((long long)N * 128), 256, 0, stream>>>(vp, vnb, N, 128);
  concat2_bf16_k<<<g1((long long)N * 256), 256, 0, stream>>>(xo, 128, vnb, 128, xcat, N);
  launch_gemm(stream, xcat, 256, b_o2u1, o2_ub1, nullptr, 0, nullptr, 0, 1.0f, 1,
              nullptr, hid, 128, N, 128, 256);
  naive_gemm_bf16A_k<<<g1((long long)N * 2), 256, 0, stream>>>(hid, 128, o2_uW2, o2_ub2, ubuf, 2, N, 2, 128, 0);
  final_vo_k<<<g1((long long)N * 3), 256, 0, stream>>>(ubuf, vec2b, outp + (long long)N * 8, N);

  (void)n_in; (void)out_size; (void)ws_size;
}
